// PCAN_29798483100273
// MI455X (gfx1250) — compile-verified, matrix operand order checked by reading, not run on hardware
//
#include <hip/hip_runtime.h>
#include <hip/hip_bf16.h>
#include <math.h>

// ---------------------------------------------------------------------------
// PCAN (PointNet++ + NetVLAD) forward for MI455X / gfx1250 (wave32).
// All matmuls run through v_wmma_f32_16x16x32_f16 with branch-free inner
// loops: operands pre-converted to f16 with K padded to 32 and leading dims
// padded to 8 halves, so each lane feeds the WMMA with two aligned
// global_load_b128 per operand per K-step. Activations live transposed
// (cols x channels) so both B-operand loads and C stores are contiguous.
// ---------------------------------------------------------------------------

typedef __attribute__((ext_vector_type(16))) _Float16 v16h;
typedef __attribute__((ext_vector_type(8)))  _Float16 v8h;
typedef __attribute__((ext_vector_type(8)))  float    v8f;
typedef __attribute__((ext_vector_type(4)))  float    v4f;

static constexpr int cB  = 4;
static constexpr int cN  = 2500;
static constexpr int cNp = 2528;    // cN padded to 32
static constexpr int cD  = 1088;    // = 34*32, no pad needed
static constexpr int cS1 = 256;
static constexpr int cCS = 64;
static constexpr int cOD = 256;
static constexpr float cEPS = 1e-5f;

// ------------------------- core WMMA GEMM (f16 ops) ------------------------
// C[col][m] (transposed, ldc = Mp) = sum_k A16[m][k] * B16[col][k]
// A16: M x lda (lda mult of 8, K zero-padded to Kp), per-lane row clamped.
// B16: Ncols x ldb (ldb mult of 8, K zero-padded), per-lane col clamped.
__global__ void k_gemm16(const _Float16* __restrict__ A16, int lda,
                         const _Float16* __restrict__ B16, int ldb,
                         float* __restrict__ C, int ldc,
                         int M, int Ncols, int Kp)
{
  const int lane = threadIdx.x;            // one wave per block
  const int r    = lane & 15;
  const int half = lane >> 4;
  const int colr = blockIdx.y * 16 + r;
  int am = blockIdx.x * 16 + r; am = (am < M) ? am : (M - 1);
  int bn = (colr < Ncols) ? colr : (Ncols - 1);
  const _Float16* ap = A16 + (long)am * lda + (half << 3);
  const _Float16* bp = B16 + (long)bn * ldb + (half << 3);
  v8f acc = {};
  for (int k0 = 0; k0 < Kp; k0 += 32) {
    v8h a0 = *(const v8h*)(ap + k0);
    v8h a1 = *(const v8h*)(ap + k0 + 16);
    v8h b0 = *(const v8h*)(bp + k0);
    v8h b1 = *(const v8h*)(bp + k0 + 16);
    v16h av = __builtin_shufflevector(a0, a1, 0,1,2,3,4,5,6,7,8,9,10,11,12,13,14,15);
    v16h bv = __builtin_shufflevector(b0, b1, 0,1,2,3,4,5,6,7,8,9,10,11,12,13,14,15);
    acc = __builtin_amdgcn_wmma_f32_16x16x32_f16(false, av, false, bv,
                                                 (short)0, acc, false, false);
  }
  if (colr < Ncols) {
    float* cp = C + (long)colr * ldc + blockIdx.x * 16 + (half << 3);
    *(v4f*)(cp)     = __builtin_shufflevector(acc, acc, 0,1,2,3);
    *(v4f*)(cp + 4) = __builtin_shufflevector(acc, acc, 4,5,6,7);
  }
}

// --------------- SA1 layer-1 GEMM with fused neighbor gather ---------------
// B column col=(b*S1+s)*Kk + k is featT16[(b*N + idx)*1088 + c] (contiguous
// along channels) for c<1088; the 3 rel-xyz channels are one extra WMMA tail
// against weight columns 1088..1090 (weights zero-padded to 1120).
__global__ void k_gemm16_sa1(const _Float16* __restrict__ W16,      // M x 1120
                             const _Float16* __restrict__ featT,    // (B*N) x 1088
                             const int* __restrict__ idx64,
                             const float* __restrict__ rel,
                             float* __restrict__ C, int ldc,
                             int M, int Ncols, int Kk)
{
  const int lane = threadIdx.x;
  const int r    = lane & 15;
  const int half = lane >> 4;
  const int colr = blockIdx.y * 16 + r;
  int am = blockIdx.x * 16 + r; am = (am < M) ? am : (M - 1);
  int cc = (colr < Ncols) ? colr : (Ncols - 1);
  const int bs  = cc / Kk;               // b*S1 + s
  const int k   = cc % Kk;
  const int b   = bs >> 8;               // / cS1
  const int nbr = idx64[bs * 64 + k];
  const _Float16* ap = W16 + (long)am * 1120 + (half << 3);
  const _Float16* bp = featT + ((long)b * cN + nbr) * cD + (half << 3);
  v8f acc = {};
  for (int k0 = 0; k0 < cD; k0 += 32) {           // 34 exact iterations
    v8h a0 = *(const v8h*)(ap + k0);
    v8h a1 = *(const v8h*)(ap + k0 + 16);
    v8h b0 = *(const v8h*)(bp + k0);
    v8h b1 = *(const v8h*)(bp + k0 + 16);
    v16h av = __builtin_shufflevector(a0, a1, 0,1,2,3,4,5,6,7,8,9,10,11,12,13,14,15);
    v16h bv = __builtin_shufflevector(b0, b1, 0,1,2,3,4,5,6,7,8,9,10,11,12,13,14,15);
    acc = __builtin_amdgcn_wmma_f32_16x16x32_f16(false, av, false, bv,
                                                 (short)0, acc, false, false);
  }
  { // rel-xyz rank-3 tail: valid K = 1088..1090 live in half==0, elems 0..2
    v8h a0 = *(const v8h*)(ap + 1088);
    v8h a1 = *(const v8h*)(ap + 1104);
    v16h av = __builtin_shufflevector(a0, a1, 0,1,2,3,4,5,6,7,8,9,10,11,12,13,14,15);
    const long rb = ((long)bs * 64 + k) * 3;
    const _Float16 z = (_Float16)0.f;
    v16h bv = {};
    bv[0] = half ? z : (_Float16)rel[rb + 0];
    bv[1] = half ? z : (_Float16)rel[rb + 1];
    bv[2] = half ? z : (_Float16)rel[rb + 2];
    acc = __builtin_amdgcn_wmma_f32_16x16x32_f16(false, av, false, bv,
                                                 (short)0, acc, false, false);
  }
  if (colr < Ncols) {
    float* cp = C + (long)colr * ldc + blockIdx.x * 16 + (half << 3);
    *(v4f*)(cp)     = __builtin_shufflevector(acc, acc, 0,1,2,3);
    *(v4f*)(cp + 4) = __builtin_shufflevector(acc, acc, 4,5,6,7);
  }
}

// ----------------------------- conversions ---------------------------------
// f32 (rows x K, ld=K) -> f16 (rows x Kp), zero pad K..Kp.
__global__ void k_cvt16(const float* __restrict__ src, _Float16* __restrict__ dst,
                        int rows, int K, int Kp)
{
  long i = (long)blockIdx.x * blockDim.x + threadIdx.x;
  if (i >= (long)rows * Kp) return;
  int r = (int)(i / Kp), k = (int)(i % Kp);
  dst[i] = (k < K) ? (_Float16)src[(long)r * K + k] : (_Float16)0.f;
}

// f32 (R x C, ld=C) -> f16 transposed (C x Rp), zero pad R..Rp.
__global__ void k_cvt16T(const float* __restrict__ src, _Float16* __restrict__ dst,
                         int R, int C, int Rp)
{
  long i = (long)blockIdx.x * blockDim.x + threadIdx.x;
  if (i >= (long)C * Rp) return;
  int c = (int)(i / Rp), r = (int)(i % Rp);
  dst[i] = (r < R) ? (_Float16)src[(long)r * C + c] : (_Float16)0.f;
}

__global__ void k_zero16(_Float16* __restrict__ p, long n)
{
  long i = (long)blockIdx.x * blockDim.x + threadIdx.x;
  if (i < n) p[i] = (_Float16)0.f;
}

// ----------------------------- KNN top-64 ----------------------------------
__global__ void k_knn64(const float* __restrict__ xyz,
                        int* __restrict__ idx64, float* __restrict__ rel)
{
  __shared__ float dist[cN];
  __shared__ float rb[256];
  __shared__ int   ri[256];
  const int q = blockIdx.x;           // b*S1 + s
  const int b = q / cS1, s = q % cS1;
  const int tid = threadIdx.x;
  const float qx = xyz[(long)b*3*cN + 0*cN + s];
  const float qy = xyz[(long)b*3*cN + 1*cN + s];
  const float qz = xyz[(long)b*3*cN + 2*cN + s];
  for (int n = tid; n < cN; n += blockDim.x) {
    float dx = xyz[(long)b*3*cN + 0*cN + n] - qx;
    float dy = xyz[(long)b*3*cN + 1*cN + n] - qy;
    float dz = xyz[(long)b*3*cN + 2*cN + n] - qz;
    dist[n] = dx*dx + dy*dy + dz*dz;
  }
  __syncthreads();
  for (int t = 0; t < 64; ++t) {
    float best = 3.4e38f; int bi = 0;
    for (int n = tid; n < cN; n += blockDim.x)
      if (dist[n] < best) { best = dist[n]; bi = n; }
    rb[tid] = best; ri[tid] = bi;
    __syncthreads();
    for (int o = 128; o; o >>= 1) {
      if (tid < o && rb[tid + o] < rb[tid]) { rb[tid] = rb[tid + o]; ri[tid] = ri[tid + o]; }
      __syncthreads();
    }
    if (tid == 0) {
      int n0 = ri[0];
      idx64[q*64 + t] = n0;
      rel[(q*64 + t)*3 + 0] = xyz[(long)b*3*cN + 0*cN + n0] - qx;
      rel[(q*64 + t)*3 + 1] = xyz[(long)b*3*cN + 1*cN + n0] - qy;
      rel[(q*64 + t)*3 + 2] = xyz[(long)b*3*cN + 2*cN + n0] - qz;
      dist[n0] = 3.4e38f;
    }
    __syncthreads();
  }
}

// -------------------- BN (batch stats) on transposed acts ------------------
// X is (cols x Mp), channel c = blockIdx.x.
__global__ void k_bn_statsT(const float* __restrict__ X, int Mp, int cols,
                            float* __restrict__ mean, float* __restrict__ var)
{
  __shared__ float ss[256], ss2[256];
  const int c = blockIdx.x, tid = threadIdx.x;
  float s = 0.f, s2 = 0.f;
  for (int n = tid; n < cols; n += blockDim.x) {
    float v = X[(long)n*Mp + c]; s += v; s2 += v*v;
  }
  ss[tid] = s; ss2[tid] = s2; __syncthreads();
  for (int o = 128; o; o >>= 1) {
    if (tid < o) { ss[tid] += ss[tid+o]; ss2[tid] += ss2[tid+o]; }
    __syncthreads();
  }
  if (tid == 0) {
    float m = ss[0] / cols;
    mean[c] = m;
    var[c]  = ss2[0] / cols - m*m;
  }
}

// normalize in place (f32) and emit f16 copy for the next GEMM's B operand.
__global__ void k_bn_applyT(float* __restrict__ X, _Float16* __restrict__ X16,
                            int Mp, int cols, int M,
                            const float* __restrict__ mean, const float* __restrict__ var,
                            const float* __restrict__ g, const float* __restrict__ bt,
                            int relu)
{
  long i = (long)blockIdx.x * blockDim.x + threadIdx.x;
  if (i >= (long)cols * Mp) return;
  int c = (int)(i % Mp);
  if (c < M) {
    float v = X[i];
    v = (v - mean[c]) * rsqrtf(var[c] + cEPS) * g[c] + bt[c];
    if (relu) v = fmaxf(v, 0.f);
    X[i] = v;
    X16[i] = (_Float16)v;
  } else {
    X16[i] = (_Float16)0.f;
  }
}

// --------------------------- pool / concat helpers -------------------------
// X (cols= (b*S1+s)*K+k  x  Mp) -> max over k -> l1p[(row_off+m)*1024 + b*S1+s]
__global__ void k_maxpool_k(const float* __restrict__ X, int K, int Mp, int M,
                            float* __restrict__ l1p, int row_off)
{
  int i = blockIdx.x * blockDim.x + threadIdx.x;     // over M * 1024
  if (i >= M * cB * cS1) return;
  int m = i / (cB*cS1), col = i % (cB*cS1);
  float mx = -3.4e38f;
  for (int k = 0; k < K; ++k)
    mx = fmaxf(mx, X[((long)col*K + k)*Mp + m]);
  l1p[(long)(row_off + m) * (cB*cS1) + col] = mx;
}

// x2t16 (1024 x 352): rows 0-2 xyz, 3-322 l1p, 323-351 zero.
__global__ void k_build_x2(const float* __restrict__ xyz,
                           const float* __restrict__ l1p, _Float16* __restrict__ x2t)
{
  int i = blockIdx.x * blockDim.x + threadIdx.x;     // 1024*352
  if (i >= cB*cS1*352) return;
  int col = i / 352, r = i % 352;
  int b = col / cS1, s = col % cS1;
  float v = 0.f;
  if (r < 3)        v = xyz[(long)b*3*cN + r*cN + s];
  else if (r < 323) v = l1p[(long)(r-3)*(cB*cS1) + col];
  x2t[i] = (_Float16)v;
}

// sa2 L3 output (1024 cols x 1024) -> per (c,b) max over s.
__global__ void k_l2max(const float* __restrict__ X, float* __restrict__ l2b)
{
  int i = blockIdx.x * blockDim.x + threadIdx.x;     // 1024 * B
  if (i >= 1024 * cB) return;
  int c = i / cB, b = i % cB;
  float mx = -3.4e38f;
  for (int s = 0; s < cS1; ++s)
    mx = fmaxf(mx, X[((long)b*cS1 + s)*1024 + c]);
  l2b[c*cB + b] = mx;
}

// xcat16 (1024 x 1344): rows 0-319 l1p, 320-1343 broadcast l2b.
__global__ void k_build_xcat(const float* __restrict__ l1p,
                             const float* __restrict__ l2b, _Float16* __restrict__ xc)
{
  int i = blockIdx.x * blockDim.x + threadIdx.x;     // 1024*1344
  if (i >= cB*cS1*1344) return;
  int col = i / 1344, r = i % 1344;
  int b = col / cS1;
  float v = (r < 320) ? l1p[(long)r*(cB*cS1) + col] : l2b[(r-320)*cB + b];
  xc[i] = (_Float16)v;
}

// ---------------------- 3-NN inverse-distance interp -----------------------
__global__ void k_nn3(const float* __restrict__ xyz,
                      int* __restrict__ idx3, float* __restrict__ w3)
{
  int i = blockIdx.x * blockDim.x + threadIdx.x;     // over B*N
  if (i >= cB * cN) return;
  int b = i / cN, n = i % cN;
  float px = xyz[(long)b*3*cN + 0*cN + n];
  float py = xyz[(long)b*3*cN + 1*cN + n];
  float pz = xyz[(long)b*3*cN + 2*cN + n];
  float d0 = 3.4e38f, d1 = 3.4e38f, d2 = 3.4e38f;
  int i0 = 0, i1 = 0, i2 = 0;
  for (int s = 0; s < cS1; ++s) {
    float dx = px - xyz[(long)b*3*cN + 0*cN + s];
    float dy = py - xyz[(long)b*3*cN + 1*cN + s];
    float dz = pz - xyz[(long)b*3*cN + 2*cN + s];
    float d = dx*dx + dy*dy + dz*dz;
    if (d < d0)      { d2=d1;i2=i1; d1=d0;i1=i0; d0=d;i0=s; }
    else if (d < d1) { d2=d1;i2=i1; d1=d;i1=s; }
    else if (d < d2) { d2=d;i2=s; }
  }
  float w0 = 1.f / (sqrtf(fmaxf(d0,0.f) + 1e-12f) + 1e-10f);
  float w1 = 1.f / (sqrtf(fmaxf(d1,0.f) + 1e-12f) + 1e-10f);
  float w2 = 1.f / (sqrtf(fmaxf(d2,0.f) + 1e-12f) + 1e-10f);
  float ws = w0 + w1 + w2;
  idx3[i*3+0] = i0; idx3[i*3+1] = i1; idx3[i*3+2] = i2;
  w3[i*3+0] = w0/ws; w3[i*3+1] = w1/ws; w3[i*3+2] = w2/ws;
}

// l1newT (1024 cols x 256 f32) -> interpT16 ((B*N) x 256 f16).
__global__ void k_interp(const float* __restrict__ l1newT,
                         const int* __restrict__ idx3, const float* __restrict__ w3,
                         _Float16* __restrict__ out16)
{
  long i = (long)blockIdx.x * blockDim.x + threadIdx.x;   // (B*N) * 256
  if (i >= (long)cB*cN*256) return;
  int col = (int)(i / 256), c = (int)(i % 256);
  int b = col / cN;
  float acc = 0.f;
#pragma unroll
  for (int j = 0; j < 3; ++j)
    acc += w3[col*3+j] * l1newT[((long)b*cS1 + idx3[col*3+j])*256 + c];
  out16[i] = (_Float16)acc;
}

// -------------------- attention + NetVLAD support kernels ------------------
// h = conv1 output (cols x 128, transposed, post-BN).
__global__ void k_conv2_att(const float* __restrict__ h, const float* __restrict__ w,
                            const float* __restrict__ bias,
                            float* __restrict__ attw, float* __restrict__ att_out)
{
  int col = blockIdx.x * blockDim.x + threadIdx.x;   // B*N
  if (col >= cB * cN) return;
  float s = bias[0];
  const float* hp = h + (long)col * 128;
  for (int r = 0; r < 128; ++r) s += w[r] * hp[r];
  float a = 1.f / (1.f + expf(-s));
  attw[col] = a;
  att_out[col] = a;
}

// act (cols x 64, transposed, post-BN): per-col softmax * att.
// Writes back f32 and scatters f16 into act16T (64 x B*cNp) for the VLAD GEMM.
__global__ void k_softmax_att(float* __restrict__ act, const float* __restrict__ attw,
                              _Float16* __restrict__ act16T)
{
  int col = blockIdx.x * blockDim.x + threadIdx.x;   // B*N
  if (col >= cB * cN) return;
  int b = col / cN, n = col % cN;
  float* base = act + (long)col * cCS;
  float mx = -3.4e38f;
  for (int k = 0; k < cCS; ++k) mx = fmaxf(mx, base[k]);
  float sum = 0.f;
  float e[cCS];
  for (int k = 0; k < cCS; ++k) { e[k] = expf(base[k] - mx); sum += e[k]; }
  float a = attw[col] / sum;
  for (int k = 0; k < cCS; ++k) {
    float v = e[k] * a;
    base[k] = v;
    act16T[(long)k*(cB*cNp) + (long)b*cNp + n] = (_Float16)v;
  }
}

__global__ void k_cluster_sums(const float* __restrict__ act, float* __restrict__ sums)
{
  __shared__ float ss[256];
  int b = blockIdx.x / cCS, k = blockIdx.x % cCS, tid = threadIdx.x;
  float s = 0.f;
  for (int n = tid; n < cN; n += blockDim.x)
    s += act[((long)b*cN + n)*cCS + k];
  ss[tid] = s; __syncthreads();
  for (int o = 128; o; o >>= 1) { if (tid < o) ss[tid] += ss[tid+o]; __syncthreads(); }
  if (tid == 0) sums[b*cCS + k] = ss[0];
}

// vladT layout: (B x CS x D), element (b,k,d) at [(b*CS + k)*D + d].
__global__ void k_vlad_sub(float* __restrict__ vladT, const float* __restrict__ sums,
                           const float* __restrict__ w2)
{
  long i = (long)blockIdx.x * blockDim.x + threadIdx.x;   // B*CS*D
  if (i >= (long)cB*cCS*cD) return;
  int b = (int)(i / (cCS*cD)), rest = (int)(i % (cCS*cD));
  int k = rest / cD, d = rest % cD;
  vladT[i] -= sums[b*cCS + k] * w2[d*cCS + k];
}

__global__ void k_vlad_norm_d(float* __restrict__ vladT)
{
  __shared__ float ss[128];
  __shared__ float nrm;
  int bk = blockIdx.x, tid = threadIdx.x;     // over B*CS
  float* base = vladT + (long)bk * cD;
  float s = 0.f;
  for (int d = tid; d < cD; d += 128) { float v = base[d]; s += v*v; }
  ss[tid] = s; __syncthreads();
  for (int o = 64; o; o >>= 1) { if (tid < o) ss[tid] += ss[tid+o]; __syncthreads(); }
  if (tid == 0) nrm = fmaxf(sqrtf(ss[0]), 1e-12f);
  __syncthreads();
  for (int d = tid; d < cD; d += 128) base[d] /= nrm;
}

__global__ void k_vlad_norm_row(float* __restrict__ vladT)
{
  __shared__ float ss[256];
  __shared__ float nrm;
  int b = blockIdx.x, tid = threadIdx.x;
  float* base = vladT + (long)b*cD*cCS;
  float s = 0.f;
  for (int i = tid; i < cD*cCS; i += 256) { float v = base[i]; s += v*v; }
  ss[tid] = s; __syncthreads();
  for (int o = 128; o; o >>= 1) { if (tid < o) ss[tid] += ss[tid+o]; __syncthreads(); }
  if (tid == 0) nrm = fmaxf(sqrtf(ss[0]), 1e-12f);
  __syncthreads();
  for (int i = tid; i < cD*cCS; i += 256) base[i] /= nrm;
}

// reorder vladT (b,k,d) -> vlad16 (b, d*CS + k) f16 to match reference flatten.
__global__ void k_cvt_vlad16(const float* __restrict__ vladT, _Float16* __restrict__ v16)
{
  long i = (long)blockIdx.x * blockDim.x + threadIdx.x;   // B * 69632
  if (i >= (long)cB*cD*cCS) return;
  int b = (int)(i / (cD*cCS)), rest = (int)(i % (cD*cCS));
  int d = rest / cCS, k = rest % cCS;
  v16[i] = (_Float16)vladT[((long)b*cCS + k)*cD + d];
}

// ------------------------------- head -------------------------------------
// Ct (256 cols x 32): rows b=0..3 valid. BN over the 4 samples per channel.
__global__ void k_bn_batch4T(const float* __restrict__ Ct, float* __restrict__ outf,
                             _Float16* __restrict__ out16,
                             const float* __restrict__ g, const float* __restrict__ bt)
{
  int c = blockIdx.x * blockDim.x + threadIdx.x;
  if (c >= cOD) return;
  float v0 = Ct[c*32+0], v1 = Ct[c*32+1], v2 = Ct[c*32+2], v3 = Ct[c*32+3];
  float m = (v0+v1+v2+v3) * 0.25f;
  float var = ((v0-m)*(v0-m) + (v1-m)*(v1-m) + (v2-m)*(v2-m) + (v3-m)*(v3-m)) * 0.25f;
  float inv = rsqrtf(var + cEPS) * g[c];
  float o0 = (v0-m)*inv + bt[c], o1 = (v1-m)*inv + bt[c];
  float o2 = (v2-m)*inv + bt[c], o3 = (v3-m)*inv + bt[c];
  outf[0*cOD+c] = o0; outf[1*cOD+c] = o1; outf[2*cOD+c] = o2; outf[3*cOD+c] = o3;
  out16[0*cOD+c] = (_Float16)o0; out16[1*cOD+c] = (_Float16)o1;
  out16[2*cOD+c] = (_Float16)o2; out16[3*cOD+c] = (_Float16)o3;
}

__global__ void k_final(const float* __restrict__ outf, const float* __restrict__ gf,
                        float* __restrict__ dst)
{
  int i = blockIdx.x * blockDim.x + threadIdx.x;
  if (i >= cB * cOD) return;
  dst[i] = outf[i] * (1.f / (1.f + expf(-gf[i])));
}

// ===========================================================================
extern "C" void kernel_launch(void* const* d_in, const int* in_sizes, int n_in,
                              void* d_out, int out_size, void* d_ws, size_t ws_size,
                              hipStream_t stream) {
  (void)in_sizes; (void)n_in; (void)out_size; (void)ws_size;
  auto F = [&](int i) { return (const float*)d_in[i]; };
  const float* xyz  = F(0);
  const float* feat = F(1);
  // params flattened in setup_inputs() insertion order (see round-0 notes):
  // sa1 2..28, sa2 29..37, fp2 38..43, fp1 44..49, conv1 50..52,
  // conv2_w 53, conv2_b 54, cluster_w 55, cluster_g 56, cluster_be 57,
  // cluster_w2 58, hidden_w 59, bn2_g 60, bn2_b 61, gate_w 62..,
  // gate_g 63, gate_b 64.

  // weight conversion table: {input idx, rows, K, Kp(=ceil32)}
  static const int WT[17][4] = {
    {2,32,1091,1120},{5,32,32,32},{8,64,32,32},
    {11,64,1091,1120},{14,64,64,64},{17,128,64,64},
    {20,64,1091,1120},{23,96,64,64},{26,128,96,96},
    {29,256,323,352},{32,512,256,256},{35,1024,512,512},
    {38,256,1344,1344},{41,256,256,256},
    {44,256,256,256},{47,128,256,256},
    {50,128,128,128}};

  // ---------------- workspace carve-up (~160 MB) ----------------
  char* wsb = (char*)d_ws;
  size_t off = 0;
  auto take = [&](size_t bytes) -> void* {
    void* p = wsb + off;
    off += (bytes + 255) & ~(size_t)255;
    return p;
  };
  int*      idx64  = (int*)     take(4ull*cB*cS1*64);
  float*    rel    = (float*)   take(4ull*cB*cS1*64*3);
  float*    bufC   = (float*)   take(4ull*65536*128);        // transposed f32 acts
  _Float16* h16A   = (_Float16*)take(2ull*65536*128);
  _Float16* h16B   = (_Float16*)take(2ull*65536*128);
  float*    l1p    = (float*)   take(4ull*320*cB*cS1);
  _Float16* x2t16  = (_Float16*)take(2ull*cB*cS1*352);
  float*    l2b    = (float*)   take(4ull*1024*cB);
  _Float16* xcat16 = (_Float16*)take(2ull*cB*cS1*1344);
  float*    l1newT = (float*)   take(4ull*cB*cS1*256);
  _Float16* l1n16  = (_Float16*)take(2ull*cB*cS1*256);       // dummy f16 sink
  int*      idx3   = (int*)     take(4ull*cB*cN*3);
  float*    w3     = (float*)   take(4ull*cB*cN*3);
  float*    attw   = (float*)   take(4ull*cB*cN);
  _Float16* act16T = (_Float16*)take(2ull*cCS*cB*cNp);
  float*    sums   = (float*)   take(4ull*cB*cCS);
  float*    vladT  = (float*)   take(4ull*cB*cCS*cD);
  _Float16* vlad16 = (_Float16*)take(2ull*cB*cD*cCS);
  float*    meanb  = (float*)   take(4ull*1024);
  float*    varb   = (float*)   take(4ull*1024);
  float*    CtO    = (float*)   take(4ull*cOD*32);
  float*    CtG    = (float*)   take(4ull*cOD*32);
  float*    outf   = (float*)   take(4ull*cB*cOD);
  float*    gatef  = (float*)   take(4ull*cB*cOD);
  _Float16* out16  = (_Float16*)take(2ull*cB*cOD);
  _Float16* gate16 = (_Float16*)take(2ull*cB*cOD);           // dummy f16 sink
  _Float16* featT16= (_Float16*)take(2ull*cB*cN*cD);         // (b,n,d)
  _Float16* feat16 = (_Float16*)take(2ull*cB*cD*cNp);        // (b,d,n) padded
  _Float16* hidwT16= (_Float16*)take(2ull*256*69632);
  _Float16* gatewT16=(_Float16*)take(2ull*256*256);
  _Float16* cwT16  = (_Float16*)take(2ull*cCS*cD);
  _Float16* w16[17];
  for (int i = 0; i < 17; ++i)
    w16[i] = (_Float16*)take(2ull*WT[i][1]*WT[i][3]);

  auto cvt = [&](const float* src, _Float16* dst, int rows, int K, int Kp) {
    long tot = (long)rows * Kp;
    k_cvt16<<<(unsigned)((tot+255)/256), 256, 0, stream>>>(src, dst, rows, K, Kp);
  };
  auto cvtT = [&](const float* src, _Float16* dst, int R, int C, int Rp) {
    long tot = (long)C * Rp;
    k_cvt16T<<<(unsigned)((tot+255)/256), 256, 0, stream>>>(src, dst, R, C, Rp);
  };
  auto gemm = [&](const _Float16* A, int lda, const _Float16* Bm, int ldb,
                  float* C, int ldc, int M, int Nc, int Kp) {
    dim3 g((M + 15)/16, (Nc + 15)/16);
    k_gemm16<<<g, 32, 0, stream>>>(A, lda, Bm, ldb, C, ldc, M, Nc, Kp);
  };
  auto bnT = [&](float* Ct, int M, int cols, const float* g, const float* b,
                 _Float16* o16, int relu) {
    k_bn_statsT<<<M, 256, 0, stream>>>(Ct, M, cols, meanb, varb);
    long tot = (long)cols * M;
    k_bn_applyT<<<(unsigned)((tot+255)/256), 256, 0, stream>>>(Ct, o16, M, cols, M,
                                                               meanb, varb, g, b, relu);
  };

  // ---------------- 0. one-time conversions ----------------
  for (int i = 0; i < 17; ++i)
    cvt(F(WT[i][0]), w16[i], WT[i][1], WT[i][2], WT[i][3]);
  cvtT(F(55), cwT16, cD, cCS, cD);            // cluster_w^T  (64 x 1088)
  cvtT(F(59), hidwT16, cD*cCS, cOD, cD*cCS);  // hidden_w^T   (256 x 69632)
  cvtT(F(62), gatewT16, cOD, cOD, cOD);       // gate_w^T     (256 x 256)
  for (int b = 0; b < cB; ++b) {
    cvt(feat + (size_t)b*cD*cN, feat16 + (size_t)b*cD*cNp, cD, cN, cNp);
    cvtT(feat + (size_t)b*cD*cN, featT16 + (size_t)b*cN*cD, cD, cN, cD);
  }

  // ---------------- 1. KNN top-64 ----------------
  k_knn64<<<cB*cS1, 256, 0, stream>>>(xyz, idx64, rel);

  // ---------------- 2. SA1 branches ----------------
  static const int sa1_co[3][3] = {{32,32,64},{64,64,128},{64,96,128}};
  static const int sa1_K[3] = {16,32,64};
  int row_off = 0;
  for (int br = 0; br < 3; ++br) {
    int K = sa1_K[br], cols = cB*cS1*K;
    int co0 = sa1_co[br][0], co1 = sa1_co[br][1], co2 = sa1_co[br][2];
    {
      dim3 g((co0+15)/16, (cols+15)/16);
      k_gemm16_sa1<<<g, 32, 0, stream>>>(w16[br*3+0], featT16, idx64, rel,
                                         bufC, co0, co0, cols, K);
    }
    bnT(bufC, co0, cols, F(WT[br*3+0][0]+1), F(WT[br*3+0][0]+2), h16A, 1);
    gemm(w16[br*3+1], co0, h16A, co0, bufC, co1, co1, cols, co0);
    bnT(bufC, co1, cols, F(WT[br*3+1][0]+1), F(WT[br*3+1][0]+2), h16B, 1);
    gemm(w16[br*3+2], co1, h16B, co1, bufC, co2, co2, cols, co1);
    bnT(bufC, co2, cols, F(WT[br*3+2][0]+1), F(WT[br*3+2][0]+2), h16A, 1);
    k_maxpool_k<<<(co2*cB*cS1 + 255)/256, 256, 0, stream>>>(bufC, K, co2, co2,
                                                            l1p, row_off);
    row_off += co2;
  }

  // ---------------- 3. SA2 global MLP + max ----------------
  k_build_x2<<<(cB*cS1*352 + 255)/256, 256, 0, stream>>>(xyz, l1p, x2t16);
  gemm(w16[9], 352, x2t16, 352, bufC, 256, 256, cB*cS1, 352);
  bnT(bufC, 256, cB*cS1, F(30), F(31), h16A, 1);
  gemm(w16[10], 256, h16A, 256, bufC, 512, 512, cB*cS1, 256);
  bnT(bufC, 512, cB*cS1, F(33), F(34), h16B, 1);
  gemm(w16[11], 512, h16B, 512, bufC, 1024, 1024, cB*cS1, 512);
  bnT(bufC, 1024, cB*cS1, F(36), F(37), h16A, 1);
  k_l2max<<<(1024*cB + 255)/256, 256, 0, stream>>>(bufC, l2b);
  k_build_xcat<<<(cB*cS1*1344 + 255)/256, 256, 0, stream>>>(l1p, l2b, xcat16);

  // ---------------- 4. FP2 ----------------
  gemm(w16[12], 1344, xcat16, 1344, bufC, 256, 256, cB*cS1, 1344);
  bnT(bufC, 256, cB*cS1, F(39), F(40), h16A, 1);
  gemm(w16[13], 256, h16A, 256, l1newT, 256, 256, cB*cS1, 256);
  bnT(l1newT, 256, cB*cS1, F(42), F(43), l1n16, 1);

  // ---------------- 5. interp -> FP1 -> conv1/conv2 ----------------
  k_nn3<<<(cB*cN + 255)/256, 256, 0, stream>>>(xyz, idx3, w3);
  k_interp<<<(unsigned)(((long)cB*cN*256 + 255)/256), 256, 0, stream>>>(l1newT, idx3,
                                                                        w3, h16A);
  gemm(w16[14], 256, h16A, 256, bufC, 256, 256, cB*cN, 256);
  bnT(bufC, 256, cB*cN, F(45), F(46), h16B, 1);
  gemm(w16[15], 256, h16B, 256, bufC, 128, 128, cB*cN, 256);
  bnT(bufC, 128, cB*cN, F(48), F(49), h16A, 1);
  gemm(w16[16], 128, h16A, 128, bufC, 128, 128, cB*cN, 128);
  bnT(bufC, 128, cB*cN, F(51), F(52), h16B, 1);
  k_conv2_att<<<(cB*cN + 255)/256, 256, 0, stream>>>(bufC, F(53), F(54), attw,
                                                     (float*)d_out + cB*cOD);

  // ---------------- 6. NetVLAD cluster ----------------
  for (int b = 0; b < cB; ++b)     // act (cols x 64) = cluster_w^T @ feat_b
    gemm(cwT16, cD, featT16 + (size_t)b*cN*cD, cD,
         bufC + (size_t)b*cN*cCS, cCS, cCS, cN, cD);
  bnT(bufC, cCS, cB*cN, F(56), F(57), h16A, 0);
  k_zero16<<<(unsigned)(((long)cCS*cB*cNp + 255)/256), 256, 0, stream>>>(
      act16T, (long)cCS*cB*cNp);
  k_softmax_att<<<(cB*cN + 255)/256, 256, 0, stream>>>(bufC, attw, act16T);
  k_cluster_sums<<<cB*cCS, 256, 0, stream>>>(bufC, sums);
  for (int b = 0; b < cB; ++b)     // vladT_b (64 x 1088) = act_b^T-weighted feat
    gemm(feat16 + (size_t)b*cD*cNp, cNp, act16T + (size_t)b*cNp, cB*cNp,
         vladT + (size_t)b*cCS*cD, cD, cD, cCS, cNp);
  k_vlad_sub<<<(unsigned)(((long)cB*cCS*cD + 255)/256), 256, 0, stream>>>(vladT, sums,
                                                                          F(58));
  k_vlad_norm_d<<<cB*cCS, 128, 0, stream>>>(vladT);
  k_vlad_norm_row<<<cB, 256, 0, stream>>>(vladT);
  k_cvt_vlad16<<<(unsigned)(((long)cB*cD*cCS + 255)/256), 256, 0, stream>>>(vladT,
                                                                            vlad16);

  // ---------------- 7. head ----------------
  gemm(vlad16, cD*cCS, hidwT16, cD*cCS, CtO, 32, cB, cOD, cD*cCS);
  k_bn_batch4T<<<1, 256, 0, stream>>>(CtO, outf, out16, F(60), F(61));
  gemm(out16, cOD, gatewT16, cOD, CtG, 32, cB, cOD, cOD);
  k_bn_batch4T<<<1, 256, 0, stream>>>(CtG, gatef, gate16, F(63), F(64));
  k_final<<<(cB*cOD + 255)/256, 256, 0, stream>>>(outf, gatef, (float*)d_out);
}
